// GatNet_64991445123454
// MI455X (gfx1250) — matrix operand order repaired and checked by reading, not compiled, not run on hardware
//
#include <hip/hip_runtime.h>
#include <hip/hip_bf16.h>

typedef __attribute__((ext_vector_type(16))) _Float16 v16h;
typedef __attribute__((ext_vector_type(8)))  _Float16 v8h;
typedef __attribute__((ext_vector_type(4)))  _Float16 v4h;
typedef __attribute__((ext_vector_type(8)))  float    v8f;

#define HEADS 8
#define CH    16
#define HC    128   // HEADS*CH
#define NEG_SLOPE 0.2f

// ---------------------------------------------------------------------------
// Utility kernels
// ---------------------------------------------------------------------------
__global__ void k_fill_f32(float* __restrict__ p, float v, long long n) {
    long long i = (long long)blockIdx.x * blockDim.x + threadIdx.x;
    if (i < n) p[i] = v;
}

// ---------------------------------------------------------------------------
// Pack W[K,128] (f32, row-major) into the exact WMMA B-fragment lane order so
// the GEMM can fetch each lane's 16 f16 values with one contiguous 32B load:
//   P[(((w*(K/32) + ks)*32) + lane)*16 + e] =
//       (f16) W[(ks*32 + (e>>3)*16 + (lane>>4)*8 + (e&7))*128 + w*16 + (lane&15)]
// ---------------------------------------------------------------------------
template <int K>
__global__ void k_pack_w(const float* __restrict__ W, _Float16* __restrict__ P) {
    int i = blockIdx.x * blockDim.x + threadIdx.x;
    if (i >= K * HC) return;
    int e    = i & 15;
    int lane = (i >> 4) & 31;
    int ks   = (i >> 9) % (K / 32);
    int w    = i / (512 * (K / 32));
    int kk   = ks * 32 + (e >> 3) * 16 + (lane >> 4) * 8 + (e & 7);
    int col  = w * 16 + (lane & 15);
    P[i] = (_Float16)W[(size_t)kk * HC + col];
}

// ---------------------------------------------------------------------------
// WMMA GEMM:  H[N, 128] = X[N, K] @ W[K, 128]   (f16 inputs, f32 accumulate)
// One block = 256 threads = 8 waves. Wave w owns output columns [16w, 16w+16)
// (i.e. head w). blockIdx.x selects a 16-row node tile. A-tile staged in LDS;
// B pre-packed in fragment order (2x global_load_b128 per k-step).
// ---------------------------------------------------------------------------
template <int K>
__global__ void k_gemm_wmma(const float* __restrict__ X, const _Float16* __restrict__ Pb,
                            float* __restrict__ Hout, int N) {
    constexpr int LDA = K + 8;                 // padded LDS row stride (f16 elems)
    __shared__ alignas(16) _Float16 xs[16 * LDA];

    const int tid  = threadIdx.x;
    const int wave = tid >> 5;
    const int lane = tid & 31;
    const int row0 = blockIdx.x * 16;

    // cooperative load of the 16 x K input tile, f32 -> f16, float4 granularity
    for (int idx = tid; idx < 16 * (K / 4); idx += 256) {
        int r  = idx / (K / 4);
        int c4 = (idx % (K / 4)) * 4;
        int gr = row0 + r;
        float4 v = make_float4(0.f, 0.f, 0.f, 0.f);
        if (gr < N) v = *(const float4*)(X + (size_t)gr * K + c4);
        v4h h4;
        h4[0] = (_Float16)v.x; h4[1] = (_Float16)v.y;
        h4[2] = (_Float16)v.z; h4[3] = (_Float16)v.w;
        *(v4h*)(&xs[r * LDA + c4]) = h4;
    }
    __syncthreads();

    const int col0  = wave * 16;
    const int nhalf = lane >> 4;   // 0: lanes 0-15, 1: lanes 16-31
    const int l16   = lane & 15;

    v8f acc = {};
    #pragma unroll
    for (int ks = 0; ks < K / 32; ++ks) {
        const int k0 = ks * 32;
        // A-fragment (16x32 f16 layout): lane l16 = row M.
        // a[0..7]  <- K = k0 + nhalf*8 + [0..7]      (16B aligned in LDS)
        // a[8..15] <- K = k0 + 16 + nhalf*8 + [0..7] (16B aligned in LDS)
        const _Float16* ap = &xs[l16 * LDA + k0 + nhalf * 8];
        v8h alo = *(const v8h*)ap;
        v8h ahi = *(const v8h*)(ap + 16);
        v16h a = __builtin_shufflevector(alo, ahi, 0, 1, 2, 3, 4, 5, 6, 7,
                                                   8, 9, 10, 11, 12, 13, 14, 15);
        // B-fragment: pre-packed, contiguous 32B per lane
        v16h b = *(const v16h*)(Pb + (((size_t)wave * (K / 32) + ks) * 32 + lane) * 16);

        acc = __builtin_amdgcn_wmma_f32_16x16x32_f16(
            /*neg_a=*/false, a, /*neg_b=*/false, b,
            /*c_mod=*/(short)0, acc, /*reuse_a=*/false, /*reuse_b=*/false);
    }

    // C/D layout: VGPR r -> row M = r (lanes 0-15) or r+8 (lanes 16-31), col = l16
    if (row0 + 16 <= N) {
        #pragma unroll
        for (int r = 0; r < 8; ++r)
            Hout[(size_t)(row0 + r + 8 * nhalf) * HC + col0 + l16] = acc[r];
    } else {
        #pragma unroll
        for (int r = 0; r < 8; ++r) {
            int gr = row0 + r + 8 * nhalf;
            if (gr < N) Hout[(size_t)gr * HC + col0 + l16] = acc[r];
        }
    }
}

// ---------------------------------------------------------------------------
// Per-node attention logits: a_src[i,h] = sum_c h[i,h,c]*att_src[h,c], same dst
// ---------------------------------------------------------------------------
__global__ void k_node_att(const float* __restrict__ Hb, const float* __restrict__ as_,
                           const float* __restrict__ ad_, float* __restrict__ asrc,
                           float* __restrict__ adst, int N) {
    int i = blockIdx.x * blockDim.x + threadIdx.x;   // node*HEADS + h
    if (i >= N * HEADS) return;
    int node = i >> 3, hh = i & 7;
    const float* hp = Hb + (size_t)node * HC + hh * CH;
    float s = 0.f, d = 0.f;
    #pragma unroll
    for (int c = 0; c < CH; ++c) {
        float v = hp[c];
        s += v * as_[hh * CH + c];
        d += v * ad_[hh * CH + c];
    }
    asrc[i] = s;
    adst[i] = d;
}

// ---------------------------------------------------------------------------
// Segment softmax over incoming edges (per destination, per head)
// ---------------------------------------------------------------------------
__device__ __forceinline__ void atomicMaxFloat(float* addr, float val) {
    if (val >= 0.0f) atomicMax((int*)addr, __float_as_int(val));
    else             atomicMin((unsigned int*)addr, __float_as_uint(val));
}

__device__ __forceinline__ float leaky(float v) {
    return v > 0.0f ? v : NEG_SLOPE * v;
}

__global__ void k_edge_max(const int* __restrict__ src, const int* __restrict__ dst,
                           const float* __restrict__ asrc, const float* __restrict__ adst,
                           float* __restrict__ m, int E, int Etot) {
    int i = blockIdx.x * blockDim.x + threadIdx.x;   // edge*HEADS + h
    if (i >= Etot * HEADS) return;
    int e = i >> 3, hh = i & 7;
    int s, d;
    if (e < E) { s = src[e]; d = dst[e]; } else { s = d = e - E; }   // self-loops
    float el = leaky(asrc[s * HEADS + hh] + adst[d * HEADS + hh]);
    atomicMaxFloat(&m[d * HEADS + hh], el);
}

__global__ void k_edge_sum(const int* __restrict__ src, const int* __restrict__ dst,
                           const float* __restrict__ asrc, const float* __restrict__ adst,
                           const float* __restrict__ m, float* __restrict__ z,
                           int E, int Etot) {
    int i = blockIdx.x * blockDim.x + threadIdx.x;
    if (i >= Etot * HEADS) return;
    int e = i >> 3, hh = i & 7;
    int s, d;
    if (e < E) { s = src[e]; d = dst[e]; } else { s = d = e - E; }
    float el = leaky(asrc[s * HEADS + hh] + adst[d * HEADS + hh]);
    atomicAdd(&z[d * HEADS + hh], __expf(el - m[d * HEADS + hh]));
}

// 32 lanes per edge: lane -> (head = lane/4, 4 channels). alpha recomputed
// (cheaper than storing 20MB of alpha; a_src/m/z are L2-resident).
__global__ void k_scatter(const int* __restrict__ src, const int* __restrict__ dst,
                          const float* __restrict__ Hb,
                          const float* __restrict__ asrc, const float* __restrict__ adst,
                          const float* __restrict__ m, const float* __restrict__ z,
                          float* __restrict__ out, int E, int Etot) {
    long long i = (long long)blockIdx.x * blockDim.x + threadIdx.x;
    if (i >= (long long)Etot * 32) return;
    int e    = (int)(i >> 5);
    int lane = (int)(i & 31);
    int hh = lane >> 2;
    int c0 = (lane & 3) * 4;
    int s, d;
    if (e < E) { s = src[e]; d = dst[e]; } else { s = d = e - E; }
    float el = leaky(asrc[s * HEADS + hh] + adst[d * HEADS + hh]);
    float alpha = __expf(el - m[d * HEADS + hh]) / (z[d * HEADS + hh] + 1e-16f);
    const float4 hv = *(const float4*)(Hb + (size_t)s * HC + hh * CH + c0);
    float* op = out + (size_t)d * HC + hh * CH + c0;
    atomicAdd(op + 0, hv.x * alpha);
    atomicAdd(op + 1, hv.y * alpha);
    atomicAdd(op + 2, hv.z * alpha);
    atomicAdd(op + 3, hv.w * alpha);
}

__global__ void k_bias_relu(float* __restrict__ x, const float* __restrict__ b, int N) {
    int i = blockIdx.x * blockDim.x + threadIdx.x;
    if (i >= N * HC) return;
    float v = x[i] + b[i & (HC - 1)];
    x[i] = v > 0.0f ? v : 0.0f;
}

// ---------------------------------------------------------------------------
// Global mean pool + MLP head
// ---------------------------------------------------------------------------
__global__ void k_pool_sum(const float* __restrict__ x, const int* __restrict__ batch,
                           float* __restrict__ sums, int N) {
    int i = blockIdx.x * blockDim.x + threadIdx.x;
    if (i >= N * HC) return;
    int node = i >> 7, c = i & (HC - 1);
    atomicAdd(&sums[(size_t)batch[node] * HC + c], x[i]);
}

__global__ void k_pool_cnt(const int* __restrict__ batch, float* __restrict__ cnt, int N) {
    int i = blockIdx.x * blockDim.x + threadIdx.x;
    if (i < N) atomicAdd(&cnt[batch[i]], 1.0f);
}

__global__ void k_mlp(const float* __restrict__ sums, const float* __restrict__ cnt,
                      const float* __restrict__ w1, const float* __restrict__ b1,
                      const float* __restrict__ w2, const float* __restrict__ b2,
                      float* __restrict__ out, int G) {
    int g = blockIdx.x * blockDim.x + threadIdx.x;
    if (g >= G) return;
    float inv = 1.0f / fmaxf(cnt[g], 1.0f);
    float o = b2[0];
    #pragma unroll
    for (int j = 0; j < 10; ++j) {
        float a = b1[j];
        for (int k = 0; k < HC; ++k) a += sums[(size_t)g * HC + k] * inv * w1[k * 10 + j];
        a = fmaxf(a, 0.0f);
        o += a * w2[j];
    }
    out[g] = o;
}

// ---------------------------------------------------------------------------
// Orchestration
// ---------------------------------------------------------------------------
static inline int cdiv(long long a, long long b) { return (int)((a + b - 1) / b); }

extern "C" void kernel_launch(void* const* d_in, const int* in_sizes, int n_in,
                              void* d_out, int out_size, void* d_ws, size_t ws_size,
                              hipStream_t stream) {
    const int F_IN = 64;
    const int N    = in_sizes[0] / F_IN;
    const int E    = in_sizes[1] / 2;
    const int Etot = E + N;           // GATConv add_self_loops=True
    const int G    = out_size;

    const float* x_in  = (const float*)d_in[0];
    const int*   ei    = (const int*)d_in[1];
    const int*   src   = ei;
    const int*   dst   = ei + E;
    const int*   batch = (const int*)d_in[2];
    const float* W[3]    = {(const float*)d_in[3],  (const float*)d_in[7],  (const float*)d_in[11]};
    const float* AS[3]   = {(const float*)d_in[4],  (const float*)d_in[8],  (const float*)d_in[12]};
    const float* AD[3]   = {(const float*)d_in[5],  (const float*)d_in[9],  (const float*)d_in[13]};
    const float* BI[3]   = {(const float*)d_in[6],  (const float*)d_in[10], (const float*)d_in[14]};
    const float* fc1_w = (const float*)d_in[15];
    const float* fc1_b = (const float*)d_in[16];
    const float* fc2_w = (const float*)d_in[17];
    const float* fc2_b = (const float*)d_in[18];

    // carve workspace
    char* ws = (char*)d_ws;
    size_t off = 0;
    auto take = [&](size_t bytes) -> void* {
        void* p = ws + off;
        off += (bytes + 255) & ~(size_t)255;
        return p;
    };
    _Float16* Wp   = (_Float16*)take((size_t)HC * HC * sizeof(_Float16));  // packed W
    float* bufA = (float*)take((size_t)N * HC * sizeof(float));   // h (projection)
    float* bufB = (float*)take((size_t)N * HC * sizeof(float));   // layer out (1,3)
    float* bufC = (float*)take((size_t)N * HC * sizeof(float));   // layer out (2)
    float* asrc = (float*)take((size_t)N * HEADS * sizeof(float));
    float* adst = (float*)take((size_t)N * HEADS * sizeof(float));
    float* mbuf = (float*)take((size_t)N * HEADS * sizeof(float));
    float* zbuf = (float*)take((size_t)N * HEADS * sizeof(float));
    float* sums = (float*)take((size_t)G * HC * sizeof(float));
    float* cnts = (float*)take((size_t)G * sizeof(float));

    const int B = 256;

    for (int l = 0; l < 3; ++l) {
        const float* xin  = (l == 0) ? x_in : (l == 1 ? bufB : bufC);
        float*       hbuf = bufA;
        float*       obuf = (l == 1) ? bufC : bufB;
        const int    K    = (l == 0) ? F_IN : HC;

        // pack weight into WMMA B-fragment order (f16)
        if (K == 64) {
            k_pack_w<64><<<cdiv(K * HC, B), B, 0, stream>>>(W[l], Wp);
            k_gemm_wmma<64><<<cdiv(N, 16), B, 0, stream>>>(xin, Wp, hbuf, N);
        } else {
            k_pack_w<128><<<cdiv(K * HC, B), B, 0, stream>>>(W[l], Wp);
            k_gemm_wmma<128><<<cdiv(N, 16), B, 0, stream>>>(xin, Wp, hbuf, N);
        }

        // per-node attention logits
        k_node_att<<<cdiv((long long)N * HEADS, B), B, 0, stream>>>(hbuf, AS[l], AD[l], asrc, adst, N);

        // segment softmax state
        k_fill_f32<<<cdiv((long long)N * HEADS, B), B, 0, stream>>>(mbuf, -3.0e38f, (long long)N * HEADS);
        k_fill_f32<<<cdiv((long long)N * HEADS, B), B, 0, stream>>>(zbuf, 0.0f, (long long)N * HEADS);
        k_fill_f32<<<cdiv((long long)N * HC, B), B, 0, stream>>>(obuf, 0.0f, (long long)N * HC);

        k_edge_max<<<cdiv((long long)Etot * HEADS, B), B, 0, stream>>>(src, dst, asrc, adst, mbuf, E, Etot);
        k_edge_sum<<<cdiv((long long)Etot * HEADS, B), B, 0, stream>>>(src, dst, asrc, adst, mbuf, zbuf, E, Etot);

        // weighted scatter-add: out[dst] += h[src] * alpha
        k_scatter<<<cdiv((long long)Etot * 32, B), B, 0, stream>>>(src, dst, hbuf, asrc, adst,
                                                                   mbuf, zbuf, obuf, E, Etot);
        // + bias, then ReLU
        k_bias_relu<<<cdiv((long long)N * HC, B), B, 0, stream>>>(obuf, BI[l], N);
    }

    // global mean pool (layer-3 output lives in bufB) + MLP head
    k_fill_f32<<<cdiv((long long)G * HC, B), B, 0, stream>>>(sums, 0.0f, (long long)G * HC);
    k_fill_f32<<<cdiv((long long)G, B), B, 0, stream>>>(cnts, 0.0f, (long long)G);
    k_pool_sum<<<cdiv((long long)N * HC, B), B, 0, stream>>>(bufB, batch, sums, N);
    k_pool_cnt<<<cdiv((long long)N, B), B, 0, stream>>>(batch, cnts, N);
    k_mlp<<<cdiv((long long)G, B), B, 0, stream>>>(sums, cnts, fc1_w, fc1_b, fc2_w, fc2_b,
                                                   (float*)d_out, G);
}
